// VisionAttention_45715631899211
// MI455X (gfx1250) — matrix operand order repaired and checked by reading, not compiled, not run on hardware
//
#include <hip/hip_runtime.h>
#include <hip/hip_bf16.h>
#include <math.h>

// ---------------- problem constants ----------------
#define LSEQ   4096
#define DMODEL 1280
#define NHEAD  16
#define HDIM   80          // head dim
#define HHALF  40          // HDIM/2 (rope)
#define QKVLD  (3*DMODEL)  // 3840, row stride of qkv buffer
#define NSEGS  4
#define NT     5           // N tiles per wave in GEMM

typedef __attribute__((ext_vector_type(16))) _Float16 v16h;
typedef __attribute__((ext_vector_type(8)))  _Float16 v8h;
typedef __attribute__((ext_vector_type(8)))  float    v8f;
typedef __attribute__((ext_vector_type(4)))  float    v4f;

#define SHUF16(a, b) __builtin_shufflevector((a), (b), 0,1,2,3,4,5,6,7,8,9,10,11,12,13,14,15)

// =====================================================================
// fp32 -> f16 bulk conversion (8 elements / thread, b128 in / b128 out)
// =====================================================================
__global__ void f32_to_f16(const float* __restrict__ in,
                           _Float16* __restrict__ out, int n8) {
  int i = blockIdx.x * blockDim.x + threadIdx.x;
  if (i >= n8) return;
  v4f a = ((const v4f*)in)[2 * i];
  v4f b = ((const v4f*)in)[2 * i + 1];
  v8h o;
#pragma unroll
  for (int j = 0; j < 4; ++j) {
    o[j]     = (_Float16)a[j];
    o[4 + j] = (_Float16)b[j];
  }
  ((v8h*)out)[i] = o;
}

// =====================================================================
// GEMM: C[M,N] = A[M,K] * W[N,K]^T + bias[N]; A,W pre-converted f16.
// One wave -> 16 x (16*NT) C strip (A fragment reused NT times).
// blockDim=256 (8 waves); grid=(M/16, N/(16*NT*8))
// =====================================================================
__global__ void gemm_bias_wmma(const _Float16* __restrict__ A,
                               const _Float16* __restrict__ W,
                               const float* __restrict__ bias,
                               float* __restrict__ C,
                               int M, int N, int K) {
  const int lane = threadIdx.x & 31;
  const int wave = threadIdx.x >> 5;
  const int m    = lane & 15;
  const int hid  = lane >> 4;
  const int mtile = blockIdx.x;
  const int nbase = (blockIdx.y * 8 + wave) * NT;   // in 16-wide tiles
  if (nbase * 16 >= N) return;

  const _Float16* arow = A + (size_t)(mtile * 16 + m) * K;
  const _Float16* wr[NT];
#pragma unroll
  for (int t = 0; t < NT; ++t)
    wr[t] = W + (size_t)((nbase + t) * 16 + m) * K;

  v8f acc[NT];
#pragma unroll
  for (int t = 0; t < NT; ++t) acc[t] = (v8f){};

  for (int kk = 0; kk < K; kk += 32) {
    // A frag: halves 0..7 = K kk+hid*8+j ; 8..15 = K kk+16+hid*8+j
    v8h a0 = *(const v8h*)(arow + kk + hid * 8);
    v8h a1 = *(const v8h*)(arow + kk + 16 + hid * 8);
    v16h af = SHUF16(a0, a1);
#pragma unroll
    for (int t = 0; t < NT; ++t) {
      // B frag: lane n=m, halves 0..7 = K kk+hid*8+j ; 8..15 = K kk+16+hid*8+j
      v8h b0 = *(const v8h*)(wr[t] + kk + hid * 8);
      v8h b1 = *(const v8h*)(wr[t] + kk + 16 + hid * 8);
      v16h bf = SHUF16(b0, b1);
      acc[t] = __builtin_amdgcn_wmma_f32_16x16x32_f16(false, af, false, bf,
                                                      (short)0, acc[t], false, false);
    }
  }

#pragma unroll
  for (int t = 0; t < NT; ++t) {
    const float bv = bias[(nbase + t) * 16 + m];
#pragma unroll
    for (int r = 0; r < 8; ++r) {
      const int orow = mtile * 16 + hid * 8 + r;    // C layout: row = hid*8+r
      C[(size_t)orow * N + (nbase + t) * 16 + m] = acc[t][r] + bv;
    }
  }
}

// =====================================================================
// RoPE in-place on q and k slices of the fp32 qkv buffer.
// =====================================================================
__global__ void rope_kernel(float* __restrict__ qkv,
                            const float* __restrict__ rope) {
  int idx = blockIdx.x * blockDim.x + threadIdx.x;
  if (idx >= LSEQ * NHEAD * HHALF) return;
  const int d = idx % HHALF;
  const int t = idx / HHALF;
  const int h = t % NHEAD;
  const int l = t / NHEAD;
  const float f = rope[(size_t)l * HHALF + d];
  const float c = __cosf(f), s = __sinf(f);
#pragma unroll
  for (int part = 0; part < 2; ++part) {   // 0 = q, 1 = k
    float* base = qkv + (size_t)l * QKVLD + part * DMODEL + h * HDIM;
    const float a = base[d];
    const float b = base[d + HHALF];
    base[d]         = a * c - b * s;
    base[d + HHALF] = b * c + a * s;
  }
}

// =====================================================================
// Block-diagonal flash attention on f16 qkv, 32 keys per iteration.
// One wave per (16-row query tile, head). grid=(LSEQ/16, NHEAD), block=32.
// V tiles staged LDS via global_load_async_to_lds_b128 (ASYNCcnt overlap).
// =====================================================================
__global__ void attn_kernel(const _Float16* __restrict__ qkvh,
                            const int* __restrict__ cu,
                            _Float16* __restrict__ O) {
  __shared__ _Float16 sP[16][32];
  __shared__ _Float16 sV[32][HDIM];

  const int lane = threadIdx.x & 31;
  const int m    = lane & 15;
  const int hid  = lane >> 4;
  const int row0 = blockIdx.x * 16;
  const int head = blockIdx.y;
  const float scale = 0.11180339887498949f;  // 1/sqrt(80)
  const v8h z8 = {};

  // ---- Q fragments: K-dim 80 padded to 96, pure b128 loads (no scaling)
  v16h qf[3];
  {
    const _Float16* qp = qkvh + (size_t)(row0 + m) * QKVLD + head * HDIM;
    v8h q0a = *(const v8h*)(qp +  0 + hid * 8);
    v8h q0b = *(const v8h*)(qp + 16 + hid * 8);
    v8h q1a = *(const v8h*)(qp + 32 + hid * 8);
    v8h q1b = *(const v8h*)(qp + 48 + hid * 8);
    v8h q2a = *(const v8h*)(qp + 64 + hid * 8);   // 64..79 valid for both halves
    qf[0] = SHUF16(q0a, q0b);
    qf[1] = SHUF16(q1a, q1b);
    qf[2] = SHUF16(q2a, z8);                      // K 80..95 zero pad
  }

  // ---- per-row segment bounds (rows in C layout: hid*8 + r)
  int rs[8], re[8];
#pragma unroll
  for (int r = 0; r < 8; ++r) {
    const int rw = row0 + hid * 8 + r;
    int s0 = 0, e0 = LSEQ;
    for (int s = 0; s < NSEGS; ++s)
      if (rw >= cu[s] && rw < cu[s + 1]) { s0 = cu[s]; e0 = cu[s + 1]; }
    rs[r] = s0; re[r] = e0;
  }
  int lo = rs[0], hi = re[0];
#pragma unroll
  for (int r = 1; r < 8; ++r) { lo = min(lo, rs[r]); hi = max(hi, re[r]); }
#pragma unroll
  for (int d = 1; d < 32; d <<= 1) {
    lo = min(lo, __shfl_xor(lo, d, 32));
    hi = max(hi, __shfl_xor(hi, d, 32));
  }
  const int kb0 = lo & ~31;

  v8f   o[5];
  float mrow[8], lrow[8];
#pragma unroll
  for (int t = 0; t < 5; ++t) o[t] = (v8f){};
#pragma unroll
  for (int r = 0; r < 8; ++r) { mrow[r] = -1e30f; lrow[r] = 0.f; }

  // per-lane constants for the async V staging: lane <-> key row 0..31
  const unsigned ldsV0 = (unsigned)(unsigned long long)(&sV[0][0]);
  const unsigned ldsRow = ldsV0 + (unsigned)lane * (HDIM * 2);

  for (int kb = kb0; kb < hi; kb += 32) {
    // ======== kick off async V-tile staging (32 keys x first 40 dims, f16) ======
    {
      const int vkey = min(kb + lane, LSEQ - 1);
      const unsigned goff =
          (unsigned)(((size_t)vkey * QKVLD + 2 * DMODEL + head * HDIM) * 2);
#pragma unroll
      for (int c = 0; c < 5; ++c) {
        unsigned l_off = ldsRow + c * 16;
        unsigned g_off = goff + c * 16;
        asm volatile("global_load_async_to_lds_b128 %0, %1, %2"
                     :: "v"(l_off), "v"(g_off), "s"(qkvh) : "memory");
      }
    }

    // ======== S = Q . K^T for 32 keys (two 16-col tiles) ========
    const int key0 = kb + m;
    const int key1 = kb + 16 + m;
    const _Float16* kp0 = qkvh + (size_t)min(key0, LSEQ - 1) * QKVLD + DMODEL + head * HDIM;
    const _Float16* kp1 = qkvh + (size_t)min(key1, LSEQ - 1) * QKVLD + DMODEL + head * HDIM;
    v8f s0 = {}, s1 = {};
#pragma unroll
    for (int half = 0; half < 2; ++half) {
      const _Float16* kp = half ? kp1 : kp0;
      v8h b0a = *(const v8h*)(kp + hid * 8);
      v8h b0b = *(const v8h*)(kp + 16 + hid * 8);
      v8h b1a = *(const v8h*)(kp + 32 + hid * 8);
      v8h b1b = *(const v8h*)(kp + 48 + hid * 8);
      v8h b2a = *(const v8h*)(kp + 64 + hid * 8);
      v8h b2b = z8;
      v16h bf0 = SHUF16(b0a, b0b);
      v16h bf1 = SHUF16(b1a, b1b);
      v16h bf2 = SHUF16(b2a, b2b);
      v8f acc = half ? s1 : s0;
      acc = __builtin_amdgcn_wmma_f32_16x16x32_f16(false, qf[0], false, bf0,
                                                   (short)0, acc, false, false);
      acc = __builtin_amdgcn_wmma_f32_16x16x32_f16(false, qf[1], false, bf1,
                                                   (short)0, acc, false, false);
      acc = __builtin_amdgcn_wmma_f32_16x16x32_f16(false, qf[2], false, bf2,
                                                   (short)0, acc, false, false);
      if (half) s1 = acc; else s0 = acc;
    }

    // ======== masked online softmax over 32 keys ========
    float alpha[8];
#pragma unroll
    for (int r = 0; r < 8; ++r) {
      float sv0 = s0[r] * scale;
      float sv1 = s1[r] * scale;
      if (key0 < rs[r] || key0 >= re[r]) sv0 = -1e30f;
      if (key1 < rs[r] || key1 >= re[r]) sv1 = -1e30f;
      float mx = fmaxf(sv0, sv1);
      mx = fmaxf(mx, __shfl_xor(mx, 1, 32));
      mx = fmaxf(mx, __shfl_xor(mx, 2, 32));
      mx = fmaxf(mx, __shfl_xor(mx, 4, 32));
      mx = fmaxf(mx, __shfl_xor(mx, 8, 32));
      const float mnew = fmaxf(mrow[r], mx);
      alpha[r] = __expf(mrow[r] - mnew);
      mrow[r]  = mnew;
      const float pv0 = (sv0 <= -1e29f) ? 0.f : __expf(sv0 - mnew);
      const float pv1 = (sv1 <= -1e29f) ? 0.f : __expf(sv1 - mnew);
      float rsum = pv0 + pv1;
      rsum += __shfl_xor(rsum, 1, 32);
      rsum += __shfl_xor(rsum, 2, 32);
      rsum += __shfl_xor(rsum, 4, 32);
      rsum += __shfl_xor(rsum, 8, 32);
      lrow[r] = lrow[r] * alpha[r] + rsum;
      // P in C layout (f16), keys 0..15 at cols 0..15, keys 16..31 at 16..31
      sP[hid * 8 + r][m]      = (_Float16)pv0;
      sP[hid * 8 + r][16 + m] = (_Float16)pv1;
    }
#pragma unroll
    for (int t = 0; t < 5; ++t)
#pragma unroll
      for (int r = 0; r < 8; ++r) o[t][r] *= alpha[r];

    asm volatile("s_wait_dscnt 0x0" ::: "memory");
    // P A-fragment: lane m, element j = key hid*16+j
    v8h pa = *(const v8h*)&sP[m][hid * 16];
    v8h pb = *(const v8h*)&sP[m][hid * 16 + 8];
    v16h pf = SHUF16(pa, pb);

    // ======== wait for async V tile, then O += P . V ========
    asm volatile("s_wait_asynccnt 0x0" ::: "memory");
#pragma unroll
    for (int t = 0; t < 5; ++t) {
      v16h vf;
#pragma unroll
      for (int j = 0; j < 16; ++j)
        vf[j] = sV[hid * 16 + j][t * 16 + m];   // B: K = hid*16+j, N = m
      o[t] = __builtin_amdgcn_wmma_f32_16x16x32_f16(false, pf, false, vf,
                                                    (short)0, o[t], false, false);
    }
  }

  // ---- normalize, write f16: O[row, head*80 + t*16 + m]
#pragma unroll
  for (int r = 0; r < 8; ++r) {
    const float inv = (lrow[r] > 0.f) ? (1.f / lrow[r]) : 0.f;
    const int orow = row0 + hid * 8 + r;
    _Float16* op = O + (size_t)orow * DMODEL + head * HDIM;
#pragma unroll
    for (int t = 0; t < 5; ++t) op[t * 16 + m] = (_Float16)(o[t][r] * inv);
  }
}

// =====================================================================
extern "C" void kernel_launch(void* const* d_in, const int* in_sizes, int n_in,
                              void* d_out, int out_size, void* d_ws, size_t ws_size,
                              hipStream_t stream) {
  const float* x     = (const float*)d_in[0];
  const int*   cu    = (const int*)  d_in[1];
  const float* rope  = (const float*)d_in[2];
  const float* Wqkv  = (const float*)d_in[3];
  const float* bqkv  = (const float*)d_in[4];
  const float* Wproj = (const float*)d_in[5];
  const float* bproj = (const float*)d_in[6];
  float* out = (float*)d_out;

  // workspace layout (all chunks 256B-aligned by construction)
  char* ws = (char*)d_ws;
  float*    qkv    = (float*)ws;                                  ws += (size_t)LSEQ * QKVLD * 4;
  _Float16* qkvh   = (_Float16*)ws;                               ws += (size_t)LSEQ * QKVLD * 2;
  _Float16* xh     = (_Float16*)ws;                               ws += (size_t)LSEQ * DMODEL * 2;
  _Float16* Wqkvh  = (_Float16*)ws;                               ws += (size_t)QKVLD * DMODEL * 2;
  _Float16* Wprojh = (_Float16*)ws;                               ws += (size_t)DMODEL * DMODEL * 2;
  _Float16* attnh  = (_Float16*)ws;

  const int n8x  = LSEQ * DMODEL / 8;
  const int n8wq = QKVLD * DMODEL / 8;
  const int n8wp = DMODEL * DMODEL / 8;
  const int n8q  = LSEQ * QKVLD / 8;

  // 0) one-time f16 conversions (L2-resident afterwards)
  f32_to_f16<<<(n8x  + 255) / 256, 256, 0, stream>>>(x,     xh,     n8x);
  f32_to_f16<<<(n8wq + 255) / 256, 256, 0, stream>>>(Wqkv,  Wqkvh,  n8wq);
  f32_to_f16<<<(n8wp + 255) / 256, 256, 0, stream>>>(Wproj, Wprojh, n8wp);

  // 1) qkv = x @ Wqkv^T + bqkv  (fp32 out for RoPE)
  gemm_bias_wmma<<<dim3(LSEQ / 16, QKVLD / (16 * NT * 8)), 256, 0, stream>>>(
      xh, Wqkvh, bqkv, qkv, LSEQ, QKVLD, DMODEL);

  // 2) RoPE on q,k in place (fp32)
  rope_kernel<<<(LSEQ * NHEAD * HHALF + 255) / 256, 256, 0, stream>>>(qkv, rope);

  // 3) qkv -> f16 for the attention core
  f32_to_f16<<<(n8q + 255) / 256, 256, 0, stream>>>(qkv, qkvh, n8q);

  // 4) block-diagonal attention (f16 in, f16 out)
  attn_kernel<<<dim3(LSEQ / 16, NHEAD), 32, 0, stream>>>(qkvh, cu, attnh);

  // 5) out = attn @ Wproj^T + bproj (fp32 out)
  gemm_bias_wmma<<<dim3(LSEQ / 16, DMODEL / (16 * NT * 8)), 256, 0, stream>>>(
      attnh, Wprojh, bproj, out, LSEQ, DMODEL, DMODEL);
}
